// LinearAttentionBlock_51281909514457
// MI455X (gfx1250) — matrix operand order repaired and checked by reading, not compiled, not run on hardware
//
#include <hip/hip_runtime.h>

// ---------------------------------------------------------------------------
// LinearAttentionBlock for MI455X (gfx1250), wave32 + V_WMMA_F32_16X16X32_BF16
// b=8, c=256, h=w=128 (n=16384), 32 groups.
// Round 3: fix TDM builtin arity (6-arg clang-23 form), keep TDM A-tile
// staging, packed b32 LDS transpose for B tiles, 2x2-tile attn GEMM waves.
// ---------------------------------------------------------------------------

typedef __attribute__((ext_vector_type(16))) __bf16        v16bf;
typedef __attribute__((ext_vector_type(8)))  float         v8f;
typedef __attribute__((ext_vector_type(4)))  unsigned int  v4u;
typedef __attribute__((ext_vector_type(2)))  unsigned int  v2u;
typedef __attribute__((ext_vector_type(4)))  float         v4f;

static constexpr int CB     = 256;    // channels
static constexpr int NN     = 16384;  // h*w
static constexpr int NBATCH = 8;
static constexpr int CPG    = 8;      // channels per group (256/32)

#define DEVI __device__ __forceinline__

union ABFrag { v4u u[2]; v16bf v; };

DEVI v8f wmma_bf16(v16bf a, v16bf b, v8f c) {
  // D = A(16x32 bf16) * B(32x16 bf16) + C(16x16 f32)
  return __builtin_amdgcn_wmma_f32_16x16x32_bf16(
      /*neg_a=*/false, a, /*neg_b=*/false, b,
      /*c_mod=*/(short)0, c, /*reuse_a=*/false, /*reuse_b=*/false);
}

DEVI unsigned short f2bf(float f) {
  unsigned int u = __float_as_uint(f);
  unsigned int r = (u + 0x7FFFu + ((u >> 16) & 1u)) >> 16;  // RNE
  return (unsigned short)r;
}

DEVI float wave_sum(float v) {
  for (int o = 16; o; o >>= 1) v += __shfl_xor(v, o, 32);
  return v;
}
DEVI float wave_max(float v) {
  for (int o = 16; o; o >>= 1) v = fmaxf(v, __shfl_xor(v, o, 32));
  return v;
}

// ---------------------------------------------------------------------------
// Tensor Data Mover: 2D tile load (Global -> LDS), bf16 elements.
// D# bit layout per CDNA5 ISA ch.8 (groups 0/1; groups 2/3 zero for 2D).
// This toolchain exposes the 6-arg builtin:
//   (uint32x4 g0, int32x8 g1, int32x4 g2, int32x4 g3, int32x8 g4, i32 cpol)
// ---------------------------------------------------------------------------
#if __has_builtin(__builtin_amdgcn_tensor_load_to_lds)
#define HAVE_TDM 1
typedef __attribute__((ext_vector_type(4))) unsigned int tdm_v4u;
typedef __attribute__((ext_vector_type(8))) int          tdm_v8i;
typedef __attribute__((ext_vector_type(4))) int          tdm_v4i;
typedef __attribute__((address_space(3))) const unsigned short lds_cus;

DEVI unsigned lds_byte_offset(const unsigned short* p) {
  return (unsigned)(unsigned long long)(lds_cus*)p;  // as(3) ptr == LDS offset
}

// tile_w elements (bf16) per row, tile_h rows; src row stride = stride_el.
// pad_int_code: pad after 2^(code+1) DWORDs; pad_amt_code: pad (code+1) DWORDs.
DEVI void tdm_load_2d_bf16(unsigned lds_off, const void* gaddr,
                           unsigned tensor_w, unsigned tensor_h,
                           unsigned stride_el, unsigned tile_w, unsigned tile_h,
                           unsigned pad_int_code, unsigned pad_amt_code,
                           bool pad_en) {
  unsigned long long ga = (unsigned long long)(size_t)gaddr;
  tdm_v4u g0;
  g0[0] = 1u;                                          // count=1 (user D#)
  g0[1] = lds_off;                                     // lds_addr
  g0[2] = (unsigned)(ga & 0xffffffffu);                // global_addr[31:0]
  g0[3] = (unsigned)((ga >> 32) & 0x1ffffffu) |        // global_addr[56:32]
          (2u << 30);                                  // type=2 ("image")
  tdm_v8i g1;
  g1[0] = (int)((1u << 16) |                           // data_size=1 (2B)
                (pad_en ? (1u << 20) : 0u) |           // pad_enable
                (pad_int_code << 22) | (pad_amt_code << 25));
  g1[1] = (int)((tensor_w & 0xffffu) << 16);           // tensor_dim0[15:0]
  g1[2] = (int)(((tensor_w >> 16) & 0xffffu) |         // tensor_dim0[31:16]
                ((tensor_h & 0xffffu) << 16));         // tensor_dim1[15:0]
  g1[3] = (int)(((tensor_h >> 16) & 0xffffu) |         // tensor_dim1[31:16]
                ((tile_w & 0xffffu) << 16));           // tile_dim0
  g1[4] = (int)(tile_h & 0xffffu);                     // tile_dim1 (tile_dim2=0)
  g1[5] = (int)stride_el;                              // tensor_dim0_stride[31:0]
  g1[6] = 0;                                           // stride hi / dim1_stride
  g1[7] = 0;
  tdm_v4i gz4 = {0, 0, 0, 0};
  tdm_v8i gz8 = {0, 0, 0, 0, 0, 0, 0, 0};
  __builtin_amdgcn_tensor_load_to_lds(g0, g1, gz4, gz4, gz8, 0);
}
#else
#define HAVE_TDM 0
#endif

// ---------------------------------------------------------------------------
// GroupNorm stats: one block per (b, group); 8*16384 contiguous floats each.
// ---------------------------------------------------------------------------
__global__ void gn_stats_kernel(const float* __restrict__ x,
                                float* __restrict__ mean,
                                float* __restrict__ rstd) {
  const int bg = blockIdx.x;                    // 0..255
  const int per = CPG * NN;                     // 131072
  const v4f* p = (const v4f*)(x + (size_t)bg * per);
  const int n4 = per / 4;
  float s = 0.f, ss = 0.f;
  for (int i = threadIdx.x; i < n4; i += blockDim.x) {
    v4f v = p[i];
    s  += v.x + v.y + v.z + v.w;
    ss += v.x * v.x + v.y * v.y + v.z * v.z + v.w * v.w;
  }
  s = wave_sum(s); ss = wave_sum(ss);
  __shared__ float red[16];
  const int wave = threadIdx.x >> 5, lane = threadIdx.x & 31;
  if (lane == 0) { red[wave] = s; red[wave + 8] = ss; }
  __syncthreads();
  if (threadIdx.x == 0) {
    float S = 0.f, SS = 0.f;
    for (int i = 0; i < 8; ++i) { S += red[i]; SS += red[i + 8]; }
    float m   = S / (float)per;
    float var = SS / (float)per - m * m;
    mean[bg] = m;
    rstd[bg] = rsqrtf(var + 1e-6f);
  }
}

// ---------------------------------------------------------------------------
// Normalize + affine, write bf16 xn [b, c, n]
// ---------------------------------------------------------------------------
__global__ void gn_norm_kernel(const float* __restrict__ x,
                               const float* __restrict__ gamma,
                               const float* __restrict__ beta,
                               const float* __restrict__ mean,
                               const float* __restrict__ rstd,
                               unsigned short* __restrict__ xn) {
  const size_t total4 = (size_t)NBATCH * CB * NN / 4;
  for (size_t i = (size_t)blockIdx.x * blockDim.x + threadIdx.x; i < total4;
       i += (size_t)gridDim.x * blockDim.x) {
    size_t f = i * 4;
    int c  = (int)((f >> 14) & 255);  // f / NN % CB
    int bg = (int)(f >> 17);          // f / (8*NN)
    v4f v = ((const v4f*)x)[i];
    float m = mean[bg], r = rstd[bg], g = gamma[c], bt = beta[c];
    float y0 = (v.x - m) * r * g + bt;
    float y1 = (v.y - m) * r * g + bt;
    float y2 = (v.z - m) * r * g + bt;
    float y3 = (v.w - m) * r * g + bt;
    v2u o;
    o.x = (unsigned)f2bf(y0) | ((unsigned)f2bf(y1) << 16);
    o.y = (unsigned)f2bf(y2) | ((unsigned)f2bf(y3) << 16);
    ((v2u*)xn)[i] = o;
  }
}

// ---------------------------------------------------------------------------
// f32 -> bf16 conversion (weights)
// ---------------------------------------------------------------------------
__global__ void cvt_bf16_kernel(const float* __restrict__ src,
                                unsigned short* __restrict__ dst, int n) {
  int i = blockIdx.x * blockDim.x + threadIdx.x;
  if (i < n) dst[i] = f2bf(src[i]);
}

// ---------------------------------------------------------------------------
// GEMM (K small): C[b, m, n] = sum_k A[m,k] * B[b,k,n]  (+bias[m]) * scale
//   M = 256, K = 256, N = 16384; A bf16 row-major; B bf16 row-major [K,N].
//   Block: 256 thr = 8 waves; tile 128(M) x 128(N); wave = 64x32 = 4x2 frags.
//   A tile staged via TDM (pad -> LDS row stride 40); B tile transposed into
//   LDS with packed b32 stores.
// ---------------------------------------------------------------------------
template <bool OUT_BF16>
__global__ void gemm_kn_kernel(const unsigned short* __restrict__ A,
                               size_t aBatchStride,
                               const unsigned short* __restrict__ Bm,
                               const float* __restrict__ bias,
                               void* __restrict__ Cm, float scale) {
  __shared__ unsigned short As[128 * 40];  // [row][k], padded stride 40
  __shared__ unsigned short Bs[128 * 40];  // [col][k] (transposed), stride 40

  const int n0 = blockIdx.x * 128;
  const int m0 = blockIdx.y * 128;
  const int b  = blockIdx.z;
  const unsigned short* Ab = A  + (size_t)b * aBatchStride;
  const unsigned short* Bb = Bm + (size_t)b * CB * NN;

  const int t    = threadIdx.x;
  const int lane = t & 31, wave = t >> 5;
  const int wy = wave >> 2, wx = wave & 3;     // 2 x 4 wave grid
  const int lm = lane & 15, lh = lane >> 4;

  const v8f vzero = {};
  v8f acc[4][2];
#pragma unroll
  for (int i = 0; i < 4; ++i)
#pragma unroll
    for (int j = 0; j < 2; ++j) acc[i][j] = vzero;

  for (int kt = 0; kt < CB / 32; ++kt) {
    const int k0 = kt * 32;
#if HAVE_TDM
    if (wave == 0) {
      // 128 rows x 32 bf16, row stride 256; LDS row stride 40 via pad:
      // pad_interval: 2^(3+1)=16 DWORDs (=32 bf16), pad_amount: 3+1=4 DWORDs.
      tdm_load_2d_bf16(lds_byte_offset(As),
                       Ab + (size_t)m0 * CB + k0,
                       /*tensor_w=*/CB - k0, /*tensor_h=*/128,
                       /*stride_el=*/CB, /*tile_w=*/32, /*tile_h=*/128,
                       /*pad_int_code=*/3, /*pad_amt_code=*/3, /*pad=*/true);
    }
#else
    {  // stage A tile: 128 rows x 32 k, 2 threads/row x 16 bf16
      int row = t >> 1, seg = t & 1;
      const unsigned short* gp = Ab + (size_t)(m0 + row) * CB + k0 + seg * 16;
      v4u u0 = *(const v4u*)gp;
      v4u u1 = *(const v4u*)(gp + 8);
      *(v4u*)&As[row * 40 + seg * 16]     = u0;
      *(v4u*)&As[row * 40 + seg * 16 + 8] = u1;
    }
#endif
    {  // stage B tile transposed: Bs[n][k] <- B[k0+k][n0+n], packed pairs
      int kp = (t & 15) * 2, ng = t >> 4;  // k pair, 8 columns per thread
      const unsigned short* gp0 = Bb + (size_t)(k0 + kp) * NN + n0 + ng * 8;
      v4u r0 = *(const v4u*)gp0;        // row kp,   n = ng*8 .. +7
      v4u r1 = *(const v4u*)(gp0 + NN); // row kp+1, n = ng*8 .. +7
      unsigned lo[4] = {r0.x, r0.y, r0.z, r0.w};
      unsigned hi[4] = {r1.x, r1.y, r1.z, r1.w};
#pragma unroll
      for (int p = 0; p < 4; ++p) {
        int n = ng * 8 + p * 2;
        unsigned a0 = (lo[p] & 0x0000ffffu) | (hi[p] << 16);
        unsigned a1 = (lo[p] >> 16)         | (hi[p] & 0xffff0000u);
        *(unsigned*)&Bs[(n    ) * 40 + kp] = a0;  // (kp,kp+1) @ col n
        *(unsigned*)&Bs[(n + 1) * 40 + kp] = a1;  // (kp,kp+1) @ col n+1
      }
    }
#if HAVE_TDM
    if (wave == 0) __builtin_amdgcn_s_wait_tensorcnt(0);
#endif
    __syncthreads();

    ABFrag af[4], bf[2];
#pragma unroll
    for (int fm = 0; fm < 4; ++fm) {  // A frag: lane(lm,lh): K = lh*8+j, 16+lh*8+j
      int row = wy * 64 + fm * 16 + lm;
      af[fm].u[0] = *(const v4u*)&As[row * 40 + lh * 8];
      af[fm].u[1] = *(const v4u*)&As[row * 40 + 16 + lh * 8];
    }
#pragma unroll
    for (int fn = 0; fn < 2; ++fn) {  // B frag: lane col=lm, K = lh*8+j, 16+lh*8+j
      int col = wx * 32 + fn * 16 + lm;
      bf[fn].u[0] = *(const v4u*)&Bs[col * 40 + lh * 8];
      bf[fn].u[1] = *(const v4u*)&Bs[col * 40 + 16 + lh * 8];
    }
#pragma unroll
    for (int fm = 0; fm < 4; ++fm)
#pragma unroll
      for (int fn = 0; fn < 2; ++fn)
        acc[fm][fn] = wmma_bf16(af[fm].v, bf[fn].v, acc[fm][fn]);
    __syncthreads();
  }

  // Epilogue: C frag lane map: VGPR r -> M = r + 8*lh, N = lm
#pragma unroll
  for (int fm = 0; fm < 4; ++fm) {
#pragma unroll
    for (int fn = 0; fn < 2; ++fn) {
      size_t col = (size_t)n0 + wx * 32 + fn * 16 + lm;
#pragma unroll
      for (int r = 0; r < 8; ++r) {
        int rowC = m0 + wy * 64 + fm * 16 + r + 8 * lh;  // channel 0..255
        float v = acc[fm][fn][r] * scale + (bias ? bias[rowC] : 0.f);
        size_t off = ((size_t)b * CB + rowC) * NN + col;
        if (OUT_BF16) ((unsigned short*)Cm)[off] = f2bf(v);
        else          ((float*)Cm)[off] = v;
      }
    }
  }
}

// ---------------------------------------------------------------------------
// Row softmax over n=16384: f32 in -> bf16 out. One block per row.
// ---------------------------------------------------------------------------
__global__ void softmax_row_kernel(const float* __restrict__ X,
                                   unsigned short* __restrict__ Y) {
  const int row = blockIdx.x;  // 0..2047 (b*c)
  const v4f* p = (const v4f*)(X + (size_t)row * NN);
  unsigned short* y = Y + (size_t)row * NN;
  __shared__ float red[8];
  const int wave = threadIdx.x >> 5, lane = threadIdx.x & 31;

  float mx = -3.4e38f;
  for (int i = threadIdx.x; i < NN / 4; i += blockDim.x) {
    v4f v = p[i];
    mx = fmaxf(mx, fmaxf(fmaxf(v.x, v.y), fmaxf(v.z, v.w)));
  }
  mx = wave_max(mx);
  if (lane == 0) red[wave] = mx;
  __syncthreads();
  float m = red[0];
  for (int i = 1; i < 8; ++i) m = fmaxf(m, red[i]);
  __syncthreads();

  float s = 0.f;
  for (int i = threadIdx.x; i < NN / 4; i += blockDim.x) {
    v4f v = p[i];
    s += __expf(v.x - m) + __expf(v.y - m) + __expf(v.z - m) + __expf(v.w - m);
  }
  s = wave_sum(s);
  if (lane == 0) red[wave] = s;
  __syncthreads();
  float tot = 0.f;
  for (int i = 0; i < 8; ++i) tot += red[i];
  float inv = 1.0f / tot;

  for (int i = threadIdx.x; i < NN / 4; i += blockDim.x) {
    v4f v = p[i];
    v2u o;
    o.x = (unsigned)f2bf(__expf(v.x - m) * inv) |
          ((unsigned)f2bf(__expf(v.y - m) * inv) << 16);
    o.y = (unsigned)f2bf(__expf(v.z - m) * inv) |
          ((unsigned)f2bf(__expf(v.w - m) * inv) << 16);
    ((v2u*)y)[i] = o;
  }
}

// ---------------------------------------------------------------------------
// attn[b,c,d] = scale * sum_n Qs[b,c,n] * Ks[b,d,n]   (K = 16384, contig-K)
// One 32x32 output block (2x2 WMMA tiles) per wave: 2x data reuse vs 16x16.
// Fragments loaded directly from global as aligned b128s.
// ---------------------------------------------------------------------------
__global__ void gemm_nk_kernel(const unsigned short* __restrict__ Qs,
                               const unsigned short* __restrict__ Ks,
                               unsigned short* __restrict__ attn, float scale) {
  const int b     = blockIdx.y;
  const int super = blockIdx.x * 8 + (threadIdx.x >> 5);  // 0..63
  const int tm = (super >> 3) * 32, tn = (super & 7) * 32;
  const int lane = threadIdx.x & 31, lm = lane & 15, lh = lane >> 4;
  const int kbase = lh * 8;

  const unsigned short* q0 = Qs + ((size_t)b * CB + tm + lm) * NN;
  const unsigned short* q1 = q0 + (size_t)16 * NN;
  const unsigned short* p0 = Ks + ((size_t)b * CB + tn + lm) * NN + lh * 8;
  const unsigned short* p1 = p0 + (size_t)16 * NN;

  v8f a00 = {}, a01 = {}, a10 = {}, a11 = {};
  for (int k0 = 0; k0 < NN; k0 += 32) {
    ABFrag fa0, fa1, fb0, fb1;
    fa0.u[0] = *(const v4u*)(q0 + k0 + kbase);
    fa0.u[1] = *(const v4u*)(q0 + k0 + 16 + kbase);
    fa1.u[0] = *(const v4u*)(q1 + k0 + kbase);
    fa1.u[1] = *(const v4u*)(q1 + k0 + 16 + kbase);
    fb0.u[0] = *(const v4u*)(p0 + k0);
    fb0.u[1] = *(const v4u*)(p0 + k0 + 16);
    fb1.u[0] = *(const v4u*)(p1 + k0);
    fb1.u[1] = *(const v4u*)(p1 + k0 + 16);
    __builtin_prefetch(q0 + k0 + 512, 0, 1);  // global_prefetch_b8
    __builtin_prefetch(q1 + k0 + 512, 0, 1);
    __builtin_prefetch(p0 + k0 + 512, 0, 1);
    __builtin_prefetch(p1 + k0 + 512, 0, 1);
    a00 = wmma_bf16(fa0.v, fb0.v, a00);
    a01 = wmma_bf16(fa0.v, fb1.v, a01);
    a10 = wmma_bf16(fa1.v, fb0.v, a10);
    a11 = wmma_bf16(fa1.v, fb1.v, a11);
  }
#pragma unroll
  for (int r = 0; r < 8; ++r) {
    int r0 = tm + r + 8 * lh, r1 = r0 + 16;
    int c0 = tn + lm,         c1 = c0 + 16;
    unsigned short* ab = attn + (size_t)b * CB * CB;
    ab[(size_t)r0 * CB + c0] = f2bf(a00[r] * scale);
    ab[(size_t)r0 * CB + c1] = f2bf(a01[r] * scale);
    ab[(size_t)r1 * CB + c0] = f2bf(a10[r] * scale);
    ab[(size_t)r1 * CB + c1] = f2bf(a11[r] * scale);
  }
}

// ---------------------------------------------------------------------------
// Host-side orchestration
// ---------------------------------------------------------------------------
extern "C" void kernel_launch(void* const* d_in, const int* in_sizes, int n_in,
                              void* d_out, int out_size, void* d_ws,
                              size_t ws_size, hipStream_t stream) {
  (void)in_sizes; (void)n_in; (void)out_size; (void)ws_size;

  const float* x     = (const float*)d_in[0];
  const float* gamma = (const float*)d_in[1];
  const float* beta  = (const float*)d_in[2];
  const float* Wq = (const float*)d_in[3];  const float* bq = (const float*)d_in[4];
  const float* Wk = (const float*)d_in[5];  const float* bk = (const float*)d_in[6];
  const float* Wv = (const float*)d_in[7];  const float* bv = (const float*)d_in[8];
  const float* Wo = (const float*)d_in[9];  const float* bo = (const float*)d_in[10];

  char* ws = (char*)d_ws;
  const size_t MB = 1ull << 20;
  float*          meanp = (float*)(ws);
  float*          rstdp = (float*)(ws + 4096);
  unsigned short* Wqb   = (unsigned short*)(ws + 8192);
  unsigned short* Wkb   = Wqb + 65536;
  unsigned short* Wvb   = Wkb + 65536;
  unsigned short* Wob   = Wvb + 65536;
  unsigned short* attn  = (unsigned short*)(ws + 1 * MB);    //  1 MB  bf16 [8,256,256]
  unsigned short* xn    = (unsigned short*)(ws + 2 * MB);    // 64 MB  bf16 [8,256,16384]
  unsigned short* Qsb   = (unsigned short*)(ws + 66 * MB);   // 64 MB
  unsigned short* Ksb   = (unsigned short*)(ws + 130 * MB);  // 64 MB
  unsigned short* Vb    = (unsigned short*)(ws + 194 * MB);  // 64 MB
  unsigned short* out2  = (unsigned short*)(ws + 258 * MB);  // 64 MB
  float*          scrF  = (float*)(ws + 322 * MB);           // 128 MB f32 (Q then K)

  // 1) GroupNorm
  gn_stats_kernel<<<256, 256, 0, stream>>>(x, meanp, rstdp);
  gn_norm_kernel<<<4096, 256, 0, stream>>>(x, gamma, beta, meanp, rstdp, xn);

  // 2) Weight conversion to bf16
  cvt_bf16_kernel<<<256, 256, 0, stream>>>(Wq, Wqb, 65536);
  cvt_bf16_kernel<<<256, 256, 0, stream>>>(Wk, Wkb, 65536);
  cvt_bf16_kernel<<<256, 256, 0, stream>>>(Wv, Wvb, 65536);
  cvt_bf16_kernel<<<256, 256, 0, stream>>>(Wo, Wob, 65536);

  // 3) Projections + softmax
  dim3 g1(NN / 128, 2, NBATCH);  // (128, 2, 8)
  gemm_kn_kernel<false><<<g1, 256, 0, stream>>>(Wqb, 0, xn, bq, scrF, 1.f);
  softmax_row_kernel<<<NBATCH * CB, 256, 0, stream>>>(scrF, Qsb);
  gemm_kn_kernel<false><<<g1, 256, 0, stream>>>(Wkb, 0, xn, bk, scrF, 1.f);
  softmax_row_kernel<<<NBATCH * CB, 256, 0, stream>>>(scrF, Ksb);
  gemm_kn_kernel<true><<<g1, 256, 0, stream>>>(Wvb, 0, xn, bv, Vb, 1.f);

  // 4) attn = softmax(Q) @ softmax(K)^T / sqrt(c)
  gemm_nk_kernel<<<dim3(8, NBATCH), 256, 0, stream>>>(Qsb, Ksb, attn, 0.0625f);

  // 5) out2 = attn @ V ; y = Wo @ out2 + bo
  gemm_kn_kernel<true><<<g1, 256, 0, stream>>>(attn, (size_t)CB * CB, Vb,
                                               nullptr, out2, 1.f);
  gemm_kn_kernel<false><<<g1, 256, 0, stream>>>(Wob, 0, out2, bo, d_out, 1.f);
}
